// SelfAttention_43207370997956
// MI455X (gfx1250) — compile-verified, matrix operand order checked by reading, not run on hardware
//
#include <hip/hip_runtime.h>

#define B_DIM 2
#define S_LEN 2048
#define EMB   1024
#define NH    16
#define HD    64
#define NEGV  (-1e20f)

typedef __attribute__((ext_vector_type(16))) __bf16 v16bf;
typedef __attribute__((ext_vector_type(8)))  float  v8f;
typedef __attribute__((ext_vector_type(8)))  unsigned short u16x8;

union AFrag {
    v16bf v;
    u16x8 h[2];
    unsigned short u[16];
};

__device__ __forceinline__ unsigned short f2bf_bits(float f) {
    union { float f; unsigned u; } x; x.f = f;
    unsigned r = x.u + 0x7FFFu + ((x.u >> 16) & 1u);
    return (unsigned short)(r >> 16);
}

__device__ __forceinline__ v8f wmma_bf16(const AFrag& a, const AFrag& b, v8f c) {
    return __builtin_amdgcn_wmma_f32_16x16x32_bf16(
        false, a.v, false, b.v, (short)0, c, false, false);
}

// A/B fragment (16x32 bf16) from row-major bf16 memory.
// Element e -> K = e + 8*grp + (e>=8 ? 8 : 0): two contiguous 16B chunks.
__device__ __forceinline__ void load_frag_bf16(AFrag& f, const unsigned short* rowptr, int grp) {
    f.h[0] = *(const u16x8*)(rowptr + 8 * grp);
    f.h[1] = *(const u16x8*)(rowptr + 16 + 8 * grp);
}

// ---------------------------------------------------------------------------
// Kernel 0: bulk fp32 -> bf16 conversion, 8 elements per thread (b128 I/O).
// ---------------------------------------------------------------------------
__global__ void __launch_bounds__(256) cvt_bf16_kernel(
    const float* __restrict__ src, unsigned short* __restrict__ dst, int n8)
{
    int i = blockIdx.x * blockDim.x + threadIdx.x;
    if (i >= n8) return;
    const float4* p = (const float4*)(src + (size_t)i * 8);
    float4 a = p[0], b = p[1];
    u16x8 o;
    o[0] = f2bf_bits(a.x); o[1] = f2bf_bits(a.y);
    o[2] = f2bf_bits(a.z); o[3] = f2bf_bits(a.w);
    o[4] = f2bf_bits(b.x); o[5] = f2bf_bits(b.y);
    o[6] = f2bf_bits(b.z); o[7] = f2bf_bits(b.w);
    *(u16x8*)(dst + (size_t)i * 8) = o;
}

// ---------------------------------------------------------------------------
// Kernel 1: Y = X @ W^T (bf16 in, bf16 out), M=4096 N=1024 K=1024.
// transposed==0 -> out[b][h][s][d]; transposed==1 -> out[b][h][d][s].
// One wave computes a 16x64 tile; inner loop software-pipelined one deep.
// ---------------------------------------------------------------------------
__global__ void __launch_bounds__(256) proj_gemm_kernel(
    const unsigned short* __restrict__ X, const unsigned short* __restrict__ W,
    unsigned short* __restrict__ out, int transposed)
{
    int wave = (blockIdx.x * blockDim.x + threadIdx.x) >> 5;
    int lane = threadIdx.x & 31;
    int grp = lane >> 4, l16 = lane & 15;
    int mt = wave & 255;          // 256 row tiles of 16
    int nb = wave >> 8;           // 16 col blocks of 64
    int row0 = mt * 16, col0 = nb * 64;

    v8f acc[4] = {};
    const unsigned short* arow = X + (size_t)(row0 + l16) * EMB;
    const unsigned short* wrow[4];
#pragma unroll
    for (int nt = 0; nt < 4; ++nt)
        wrow[nt] = W + (size_t)(col0 + nt * 16 + l16) * EMB;

    // prologue: load k-chunk 0
    AFrag aCur, bCur[4];
    load_frag_bf16(aCur, arow, grp);
#pragma unroll
    for (int nt = 0; nt < 4; ++nt)
        load_frag_bf16(bCur[nt], wrow[nt], grp);

#pragma unroll 2
    for (int k0 = 0; k0 < EMB - 32; k0 += 32) {
        // issue next iteration's loads before this iteration's WMMAs
        AFrag aNxt, bNxt[4];
        load_frag_bf16(aNxt, arow + k0 + 32, grp);
#pragma unroll
        for (int nt = 0; nt < 4; ++nt)
            load_frag_bf16(bNxt[nt], wrow[nt] + k0 + 32, grp);
#pragma unroll
        for (int nt = 0; nt < 4; ++nt)
            acc[nt] = wmma_bf16(aCur, bCur[nt], acc[nt]);
        aCur = aNxt;
#pragma unroll
        for (int nt = 0; nt < 4; ++nt)
            bCur[nt] = bNxt[nt];
    }
#pragma unroll
    for (int nt = 0; nt < 4; ++nt)
        acc[nt] = wmma_bf16(aCur, bCur[nt], acc[nt]);

#pragma unroll
    for (int nt = 0; nt < 4; ++nt) {
        int col = col0 + nt * 16 + l16;
        int h = col >> 6, d = col & 63;
#pragma unroll
        for (int r = 0; r < 8; ++r) {
            int m  = row0 + r + 8 * grp;       // global row = b*S + s
            int b_ = m >> 11, s = m & (S_LEN - 1);
            size_t idx = transposed
                ? ((size_t)(b_ * NH + h) * HD + d) * S_LEN + s
                : ((size_t)(b_ * NH + h) * S_LEN + s) * HD + d;
            out[idx] = f2bf_bits(acc[nt][r]);
        }
    }
}

// ---------------------------------------------------------------------------
// Kernel 2: flash attention. One wave = one (b,h) x 16 query rows.
// Q,K: [b][h][s][d] bf16.  Vt: [b][h][d][s] bf16.  mask: [S][S] int32.
// ---------------------------------------------------------------------------
__global__ void __launch_bounds__(256) attn_kernel(
    const unsigned short* __restrict__ Qb, const unsigned short* __restrict__ Kb,
    const unsigned short* __restrict__ Vt, const int* __restrict__ mask,
    unsigned short* __restrict__ attnOut)
{
    __shared__ unsigned short Pst[8 * 16 * 32];   // per-wave 16x32 P staging

    int wave = (blockIdx.x * blockDim.x + threadIdx.x) >> 5;
    int wib  = threadIdx.x >> 5;
    int lane = threadIdx.x & 31;
    int grp = lane >> 4, l16 = lane & 15;

    int bh = wave >> 7;            // 32 (b,h) pairs
    int qt = wave & 127;           // 128 q tiles of 16
    int q0 = qt * 16;

    const unsigned short* Qh = Qb + (size_t)bh * S_LEN * HD;
    const unsigned short* Kh = Kb + (size_t)bh * S_LEN * HD;
    const unsigned short* Vh = Vt + (size_t)bh * HD * S_LEN;
    unsigned short* Pw = Pst + wib * (16 * 32);

    AFrag aq0, aq1;
    load_frag_bf16(aq0, Qh + (size_t)(q0 + l16) * HD,      grp);
    load_frag_bf16(aq1, Qh + (size_t)(q0 + l16) * HD + 32, grp);

    const int* mrow[8];
#pragma unroll
    for (int r = 0; r < 8; ++r)
        mrow[r] = mask + (size_t)(q0 + r + 8 * grp) * S_LEN;

    float mMax[8], lSum[8];
#pragma unroll
    for (int r = 0; r < 8; ++r) { mMax[r] = -3.0e38f; lSum[r] = 0.f; }
    v8f acc[4] = {};

    const float scale = 0.03125f;   // 1/sqrt(1024)

    for (int kt = 0; kt < S_LEN; kt += 32) {
        // ---- hoist independent loads: mask bits for this 16x32 block ----
        int mv0[8], mv1[8];
#pragma unroll
        for (int r = 0; r < 8; ++r) {
            mv0[r] = mrow[r][kt + l16];
            mv1[r] = mrow[r][kt + 16 + l16];
        }

        // ---- scores: two 16x16 tiles (cols kt..kt+15, kt+16..kt+31) ----
        AFrag bk;
        v8f s0 = {};
        load_frag_bf16(bk, Kh + (size_t)(kt + l16) * HD,      grp);
        s0 = wmma_bf16(aq0, bk, s0);
        load_frag_bf16(bk, Kh + (size_t)(kt + l16) * HD + 32, grp);
        s0 = wmma_bf16(aq1, bk, s0);
        v8f s1 = {};
        load_frag_bf16(bk, Kh + (size_t)(kt + 16 + l16) * HD,      grp);
        s1 = wmma_bf16(aq0, bk, s1);
        load_frag_bf16(bk, Kh + (size_t)(kt + 16 + l16) * HD + 32, grp);
        s1 = wmma_bf16(aq1, bk, s1);

        // ---- issue V fragment loads early; softmax VALU hides their latency
        AFrag bv[4];
#pragma unroll
        for (int nt = 0; nt < 4; ++nt)
            load_frag_bf16(bv[nt], Vh + (size_t)(nt * 16 + l16) * S_LEN + kt, grp);

        // ---- prefetch next K tile into cache (global_prefetch_b8) ----
        if (kt + 32 < S_LEN) {
            __builtin_prefetch(Kh + (size_t)(kt + 32 + l16) * HD, 0, 3);
            __builtin_prefetch(Kh + (size_t)(kt + 48 + l16) * HD, 0, 3);
        }

        // ---- mask + online softmax (row stats via 16-lane xor reduce) ----
        float p0[8], p1[8], corr[8];
#pragma unroll
        for (int r = 0; r < 8; ++r) {
            float v0 = s0[r] * scale;
            float v1 = s1[r] * scale;
            if (mv0[r] == 0) v0 = NEGV;
            if (mv1[r] == 0) v1 = NEGV;
            float tmax = fmaxf(v0, v1);
#pragma unroll
            for (int off = 1; off < 16; off <<= 1)
                tmax = fmaxf(tmax, __shfl_xor(tmax, off, 32));
            float mNew = fmaxf(mMax[r], tmax);
            float c = __expf(mMax[r] - mNew);
            v0 = __expf(v0 - mNew);
            v1 = __expf(v1 - mNew);
            float ts = v0 + v1;
#pragma unroll
            for (int off = 1; off < 16; off <<= 1)
                ts += __shfl_xor(ts, off, 32);
            lSum[r] = lSum[r] * c + ts;
            mMax[r] = mNew;
            corr[r] = c;
            p0[r] = v0; p1[r] = v1;
        }
#pragma unroll
        for (int nt = 0; nt < 4; ++nt)
#pragma unroll
            for (int r = 0; r < 8; ++r)
                acc[nt][r] *= corr[r];

        // ---- C-layout -> A-layout transpose of P through LDS ----
#pragma unroll
        for (int r = 0; r < 8; ++r) {
            int rowm = r + 8 * grp;
            Pw[rowm * 32 + l16]      = f2bf_bits(p0[r]);
            Pw[rowm * 32 + 16 + l16] = f2bf_bits(p1[r]);
        }
        __asm__ volatile("s_wait_dscnt 0x0" ::: "memory");
        AFrag ap;
        {
            const unsigned short* prow = Pw + l16 * 32;
            ap.h[0] = *(const u16x8*)(prow + 8 * grp);
            ap.h[1] = *(const u16x8*)(prow + 16 + 8 * grp);
        }

        // ---- acc += P(16x32) @ V(32x64) ----
#pragma unroll
        for (int nt = 0; nt < 4; ++nt)
            acc[nt] = wmma_bf16(ap, bv[nt], acc[nt]);
    }

    // ---- normalize + store to [b][s][h*64+d] bf16 ----
    int b_ = bh >> 4, h = bh & 15;
#pragma unroll
    for (int r = 0; r < 8; ++r) {
        float inv = lSum[r] > 0.f ? 1.f / lSum[r] : 0.f;
        int s = q0 + r + 8 * grp;
#pragma unroll
        for (int nt = 0; nt < 4; ++nt) {
            int d = nt * 16 + l16;
            size_t idx = (size_t)(b_ * S_LEN + s) * EMB + h * HD + d;
            attnOut[idx] = f2bf_bits(acc[nt][r] * inv);
        }
    }
}

// ---------------------------------------------------------------------------
// Kernel 3: out = A @ Wo^T + bo (A, Wo bf16; bo fp32; out fp32).
// ---------------------------------------------------------------------------
__global__ void __launch_bounds__(256) out_proj_kernel(
    const unsigned short* __restrict__ A, const unsigned short* __restrict__ W,
    const float* __restrict__ bias, float* __restrict__ out)
{
    int wave = (blockIdx.x * blockDim.x + threadIdx.x) >> 5;
    int lane = threadIdx.x & 31;
    int grp = lane >> 4, l16 = lane & 15;
    int mt = wave & 255;
    int nb = wave >> 8;
    int row0 = mt * 16, col0 = nb * 64;

    v8f acc[4] = {};
    const unsigned short* arow = A + (size_t)(row0 + l16) * EMB;
    const unsigned short* wrow[4];
#pragma unroll
    for (int nt = 0; nt < 4; ++nt)
        wrow[nt] = W + (size_t)(col0 + nt * 16 + l16) * EMB;

    AFrag aCur, bCur[4];
    load_frag_bf16(aCur, arow, grp);
#pragma unroll
    for (int nt = 0; nt < 4; ++nt)
        load_frag_bf16(bCur[nt], wrow[nt], grp);

#pragma unroll 2
    for (int k0 = 0; k0 < EMB - 32; k0 += 32) {
        AFrag aNxt, bNxt[4];
        load_frag_bf16(aNxt, arow + k0 + 32, grp);
#pragma unroll
        for (int nt = 0; nt < 4; ++nt)
            load_frag_bf16(bNxt[nt], wrow[nt] + k0 + 32, grp);
#pragma unroll
        for (int nt = 0; nt < 4; ++nt)
            acc[nt] = wmma_bf16(aCur, bCur[nt], acc[nt]);
        aCur = aNxt;
#pragma unroll
        for (int nt = 0; nt < 4; ++nt)
            bCur[nt] = bNxt[nt];
    }
#pragma unroll
    for (int nt = 0; nt < 4; ++nt)
        acc[nt] = wmma_bf16(aCur, bCur[nt], acc[nt]);

#pragma unroll
    for (int nt = 0; nt < 4; ++nt) {
        int col = col0 + nt * 16 + l16;
        float bo = bias[col];
#pragma unroll
        for (int r = 0; r < 8; ++r) {
            int m = row0 + r + 8 * grp;
            out[(size_t)m * EMB + col] = acc[nt][r] + bo;
        }
    }
}

// ---------------------------------------------------------------------------
extern "C" void kernel_launch(void* const* d_in, const int* in_sizes, int n_in,
                              void* d_out, int out_size, void* d_ws, size_t ws_size,
                              hipStream_t stream) {
    (void)in_sizes; (void)n_in; (void)out_size; (void)ws_size;

    const float* query  = (const float*)d_in[0];
    const float* keys   = (const float*)d_in[1];
    const float* values = (const float*)d_in[2];
    const int*   mask   = (const int*)  d_in[3];
    const float* Wq     = (const float*)d_in[4];
    const float* Wk     = (const float*)d_in[5];
    const float* Wv     = (const float*)d_in[6];
    const float* Wo     = (const float*)d_in[7];
    const float* bo     = (const float*)d_in[8];
    float* out = (float*)d_out;

    const size_t ACT = (size_t)B_DIM * S_LEN * EMB;   // 4194304 elems
    const size_t WEL = (size_t)EMB * EMB;             // 1048576 elems
    unsigned short* Qbf = (unsigned short*)d_ws;      // head-layout Q
    unsigned short* Kbf = Qbf + ACT;                  // head-layout K
    unsigned short* Vtb = Kbf + ACT;                  // head-layout V^T
    unsigned short* Abf = Vtb + ACT;                  // attention output
    unsigned short* Xq  = Abf + ACT;                  // bf16 activations
    unsigned short* Xk  = Xq  + ACT;
    unsigned short* Xv  = Xk  + ACT;
    unsigned short* Wqb = Xv  + ACT;                  // bf16 weights
    unsigned short* Wkb = Wqb + WEL;
    unsigned short* Wvb = Wkb + WEL;
    unsigned short* Wob = Wvb + WEL;

    dim3 block(256);
    int actBlocks = (int)(ACT / 8 / 256);   // 2048
    int wBlocks   = (int)(WEL / 8 / 256);   // 512
    cvt_bf16_kernel<<<actBlocks, block, 0, stream>>>(query,  Xq,  (int)(ACT / 8));
    cvt_bf16_kernel<<<actBlocks, block, 0, stream>>>(keys,   Xk,  (int)(ACT / 8));
    cvt_bf16_kernel<<<actBlocks, block, 0, stream>>>(values, Xv,  (int)(ACT / 8));
    cvt_bf16_kernel<<<wBlocks,   block, 0, stream>>>(Wq,     Wqb, (int)(WEL / 8));
    cvt_bf16_kernel<<<wBlocks,   block, 0, stream>>>(Wk,     Wkb, (int)(WEL / 8));
    cvt_bf16_kernel<<<wBlocks,   block, 0, stream>>>(Wv,     Wvb, (int)(WEL / 8));
    cvt_bf16_kernel<<<wBlocks,   block, 0, stream>>>(Wo,     Wob, (int)(WEL / 8));

    dim3 grid(512);
    proj_gemm_kernel<<<grid, block, 0, stream>>>(Xq, Wqb, Qbf, 0);
    proj_gemm_kernel<<<grid, block, 0, stream>>>(Xk, Wkb, Kbf, 0);
    proj_gemm_kernel<<<grid, block, 0, stream>>>(Xv, Wvb, Vtb, 1);
    attn_kernel<<<grid, block, 0, stream>>>(Qbf, Kbf, Vtb, mask, Abf);
    out_proj_kernel<<<grid, block, 0, stream>>>(Abf, Wob, bo, out);
}
